// snowball_rand_49022756716634
// MI455X (gfx1250) — compile-verified, matrix operand order checked by reading, not run on hardware
//
#include <hip/hip_runtime.h>
#include <hip/hip_bf16.h>
#include <math.h>
#include <stdint.h>

typedef __bf16 bf16;
typedef __attribute__((ext_vector_type(16))) __bf16 v16bf;
typedef __attribute__((ext_vector_type(8)))  float  v8f;
typedef int v4i_g __attribute__((vector_size(16)));   // matches builtin pointee type

// ---------------- problem constants ----------------
#define NFEAT   512
#define NHID    256
#define NLAYERS 4
#define NCLASS  40
#define CATDIM  (NFEAT + NLAYERS * NHID)   // 1536

// ---------------- gfx1250 async global->LDS support ----------------
#if defined(__has_builtin)
#if __has_builtin(__builtin_amdgcn_global_load_async_to_lds_b128)
#define ATHENA_ASYNC_LDS 1
#endif
#endif

__device__ __forceinline__ void async_copy16(const bf16* g, bf16* l) {
#ifdef ATHENA_ASYNC_LDS
  __builtin_amdgcn_global_load_async_to_lds_b128(
      (__attribute__((address_space(1))) v4i_g*)(uintptr_t)g,
      (__attribute__((address_space(3))) v4i_g*)(uintptr_t)l,
      /*offset=*/0, /*cpol=*/0);
#else
  *(uint4*)l = *(const uint4*)g;
#endif
}

template <short NREM>
__device__ __forceinline__ void wait_async_le() {
#ifdef ATHENA_ASYNC_LDS
#if __has_builtin(__builtin_amdgcn_s_wait_asynccnt)
  __builtin_amdgcn_s_wait_asynccnt(NREM);
#else
  asm volatile("s_wait_asynccnt %0" :: "i"(NREM) : "memory");
#endif
#endif
}

// ---------------- small utility kernels ----------------

// x [N,512] f32 -> XB[:, 0:512] bf16 with row stride CATDIM
__global__ void k_cast_x(const float* __restrict__ x,
                         bf16* __restrict__ XB, int N) {
  long long i = (long long)blockIdx.x * blockDim.x + threadIdx.x;
  long long total = (long long)N * NFEAT;
  if (i >= total) return;
  int row = (int)(i / NFEAT);
  int col = (int)(i - (long long)row * NFEAT);
  XB[(long long)row * CATDIM + col] = (bf16)x[i];
}

__global__ void k_zero_f32(float* __restrict__ p, long long n) {
  long long i = (long long)blockIdx.x * blockDim.x + threadIdx.x;
  if (i < n) p[i] = 0.0f;
}

// ---------------- weight pack: W[K,nout] f32 -> WMMA B-fragment layout bf16 ----
// Bp[((kstep*ntp + nt)*32 + lane)*16 + i]
//   lane<16 : elems 0..7 -> K 0..7,  elems 8..15 -> K 16..23 (col n = nt*16+lane)
//   lane>=16: elems 0..7 -> K 8..15, elems 8..15 -> K 24..31 (col n = nt*16+lane-16)
__global__ void k_pack_w(const float* __restrict__ W, int ldw, int K, int nout,
                         int ntp, bf16* __restrict__ Bp, long long total) {
  long long gid = (long long)blockIdx.x * blockDim.x + threadIdx.x;
  if (gid >= total) return;
  int i     = (int)(gid & 15);
  int lane  = (int)((gid >> 4) & 31);
  long long t = gid >> 9;
  int nt    = (int)(t % ntp);
  int kstep = (int)(t / ntp);
  int hs    = lane >> 4;
  int n     = nt * 16 + (lane & 15);
  int koff  = ((i < 8) ? i : i + 8) + hs * 8;
  int k     = kstep * 32 + koff;
  bf16 v = (bf16)0.0f;
  if (n < nout && k < K) v = (bf16)W[(long long)k * ldw + n];
  Bp[gid] = v;
}

// ---------------- WMMA bf16 GEMM: H[M,nout] = A[M,0:K] @ W ----------------
// block 256 thr = 8 waves; block tile 128(M) x 64(N); wave: 16 rows x 4 n-tiles.
// A: triple-buffered LDS tile 128x32 staged with global_load_async_to_lds_b128;
//    prefetch of tile s+1 issued BEFORE the barrier and left in flight through
//    the WMMAs (s_wait_asynccnt 2). Buf (s+1)%3 was last read at iter s-2,
//    whose reads were consumed before that iteration's barrier signal -> safe.
// B: pre-packed fragments, coalesced global b128 loads (L2-resident).
#define LDSPAD 48

union FragU { v16bf v; uint4 q[2]; };

__global__ __launch_bounds__(256)
void k_gemm_wmma_bf16(const bf16* __restrict__ A, int lda,
                      const bf16* __restrict__ Bp, int ntp,
                      float* __restrict__ H, int ldh,
                      int M, int K, int nout) {
  __shared__ bf16 As[3][128][LDSPAD];

  const int tid  = threadIdx.x;
  const int lane = tid & 31;
  const int wave = tid >> 5;
  const int l16  = lane & 15;
  const int hs   = (lane >= 16) ? 1 : 0;
  const int row0 = blockIdx.x * 128;
  const int col0 = blockIdx.y * 64;

  // --- staging assignment: thread -> 16 contiguous bf16 (32B) of the A tile ---
  const int sr  = tid >> 1;            // tile row 0..127
  const int skk = (tid & 1) * 16;      // K elem offset 0 or 16
  const bool srok = (row0 + sr) < M;
  if (!srok) {                         // zero OOB rows once; async never touches them
    const uint4 zq = {0u, 0u, 0u, 0u};
    #pragma unroll
    for (int b = 0; b < 3; ++b) {
      *(uint4*)(&As[b][sr][skk])     = zq;
      *(uint4*)(&As[b][sr][skk + 8]) = zq;
    }
  }
  const bf16* gsrc = A + (long long)(row0 + sr) * lda + skk;

  // prologue: tile 0 -> buf 0
  if (srok) {
    async_copy16(gsrc,     &As[0][sr][skk]);
    async_copy16(gsrc + 8, &As[0][sr][skk + 8]);
  }

  v8f acc[4] = {{}, {}, {}, {}};
  const int ksteps = K >> 5;
  const int ntbase = blockIdx.y * 4;

  int buf = 0, nbuf = 1;
  for (int s = 0; s < ksteps; ++s) {
    // issue prefetch of tile s+1, then wait only for tile s (prefetch stays in flight)
    if (s + 1 < ksteps) {
      if (srok) {
        const bf16* g = gsrc + (long long)(s + 1) * 32;
        async_copy16(g,     &As[nbuf][sr][skk]);
        async_copy16(g + 8, &As[nbuf][sr][skk + 8]);
      }
      wait_async_le<2>();
    } else {
      wait_async_le<0>();
    }
    __syncthreads();

    // A fragment (row-per-lane): two ds_load_b128
    FragU fa;
    {
      const int ar = wave * 16 + l16;
      const int ko = hs * 8;
      fa.q[0] = *(const uint4*)(&As[buf][ar][ko]);
      fa.q[1] = *(const uint4*)(&As[buf][ar][ko + 16]);
    }
    // B fragments: packed, 32B contiguous per lane, coalesced
    FragU fb[4];
    {
      const bf16* bp = Bp + ((long long)(s * ntp + ntbase) * 32 + lane) * 16;
      #pragma unroll
      for (int j = 0; j < 4; ++j) {
        fb[j].q[0] = *(const uint4*)(bp + j * 512);
        fb[j].q[1] = *(const uint4*)(bp + j * 512 + 8);
      }
    }

    #pragma unroll
    for (int j = 0; j < 4; ++j) {
      acc[j] = __builtin_amdgcn_wmma_f32_16x16x32_bf16(
          false, fa.v, false, fb[j].v, (short)0, acc[j], false, false);
    }

    buf = nbuf;
    nbuf = nbuf + 1; if (nbuf == 3) nbuf = 0;
  }

  // --- store C: VGPR r -> M = hs*8 + r, N = lane%16 ---
  #pragma unroll
  for (int j = 0; j < 4; ++j) {
    const int col = col0 + j * 16 + l16;
    if (col < nout) {
      #pragma unroll
      for (int r = 0; r < 8; ++r) {
        const int row = row0 + wave * 16 + hs * 8 + r;
        if (row < M) H[(long long)row * ldh + col] = acc[j][r];
      }
    }
  }
}

// ---------------- SpMM: Y[dst] += val * Hs[src]  (float4 gather + f32 atomics) ----
__global__ void k_spmm(const float* __restrict__ Hs, int ldh,
                       const float* __restrict__ eval,
                       const int* __restrict__ esrc,
                       const int* __restrict__ edst,
                       float* __restrict__ Y, int ldy,
                       int F4, long long total) {
  long long gid = (long long)blockIdx.x * blockDim.x + threadIdx.x;
  if (gid >= total) return;
  long long e = gid / F4;
  int q = (int)(gid - e * F4);
  int f = q * 4;
  int s = esrc[e];
  int d = edst[e];
  float v = eval[e];
  const float4 h = *(const float4*)(&Hs[(long long)s * ldh + f]);
  float* yb = &Y[(long long)d * ldy + f];
  atomicAdd(yb + 0, h.x * v);
  atomicAdd(yb + 1, h.y * v);
  atomicAdd(yb + 2, h.z * v);
  atomicAdd(yb + 3, h.w * v);
}

// ---------------- tanh(AH + b) -> append into XB (bf16) ----------------
__global__ void k_bias_tanh_append(const float* __restrict__ AH,
                                   const float* __restrict__ b,
                                   bf16* __restrict__ XB, int colofs, int N) {
  long long i = (long long)blockIdx.x * blockDim.x + threadIdx.x;
  long long total = (long long)N * NHID;
  if (i >= total) return;
  int row = (int)(i / NHID);
  int f   = (int)(i - (long long)row * NHID);
  float h = tanhf(AH[i] + b[f]);
  XB[(long long)row * CATDIM + colofs + f] = (bf16)h;
}

// ---------------- final: log_softmax(OUT + b_out) -> d_out ----------------
__global__ void k_bias_logsoftmax(const float* __restrict__ OUT,
                                  const float* __restrict__ b,
                                  float* __restrict__ out, int N) {
  int n = blockIdx.x * blockDim.x + threadIdx.x;
  if (n >= N) return;
  float v[NCLASS];
  float mx = -3.4e38f;
  #pragma unroll
  for (int c = 0; c < NCLASS; ++c) {
    v[c] = OUT[(long long)n * NCLASS + c] + b[c];
    mx = fmaxf(mx, v[c]);
  }
  float s = 0.0f;
  #pragma unroll
  for (int c = 0; c < NCLASS; ++c) s += expf(v[c] - mx);
  float ls = logf(s);
  #pragma unroll
  for (int c = 0; c < NCLASS; ++c)
    out[(long long)n * NCLASS + c] = v[c] - mx - ls;
}

// ---------------- host orchestration ----------------

static inline size_t alignup(size_t x) { return (x + 255) & ~(size_t)255; }

extern "C" void kernel_launch(void* const* d_in, const int* in_sizes, int n_in,
                              void* d_out, int out_size, void* d_ws, size_t ws_size,
                              hipStream_t stream) {
  const float* x        = (const float*)d_in[0];
  const float* edge_val = (const float*)d_in[1];
  const int*   edge_src = (const int*)d_in[2];
  const int*   edge_dst = (const int*)d_in[3];
  const float* Wk[NLAYERS] = {(const float*)d_in[4], (const float*)d_in[6],
                              (const float*)d_in[8], (const float*)d_in[10]};
  const float* bk[NLAYERS] = {(const float*)d_in[5], (const float*)d_in[7],
                              (const float*)d_in[9], (const float*)d_in[11]};
  const float* W_out = (const float*)d_in[12];
  const float* b_out = (const float*)d_in[13];

  const int N = in_sizes[0] / NFEAT;          // 50000
  const long long E = in_sizes[1];            // 1,600,000

  const int NTP_HID = ((NHID + 63) / 64) * 4;   // 16 packed n-tiles
  const int NTP_OUT = ((NCLASS + 63) / 64) * 4; // 4 packed n-tiles (tiles >=3 zero)

  // ---- workspace layout ----
  char* ws = (char*)d_ws;
  size_t off = 0;
  bf16* XB = (bf16*)(ws + off); off = alignup(off + (size_t)N * CATDIM * sizeof(bf16));
  float* H  = (float*)(ws + off); off = alignup(off + (size_t)N * NHID * sizeof(float));
  float* AH = (float*)(ws + off); off = alignup(off + (size_t)N * NHID * sizeof(float));
  // packed weights
  long long bofs[NLAYERS];
  long long btot = 0;
  for (int k = 0; k < NLAYERS; ++k) {
    bofs[k] = btot;
    long long fan_in = NFEAT + (long long)k * NHID;
    btot += (fan_in / 32) * NTP_HID * 512;     // ksteps * ntp * 32lanes * 16elems
  }
  bf16* BpH = (bf16*)(ws + off); off = alignup(off + (size_t)btot * sizeof(bf16));
  long long btot_out = ((long long)CATDIM / 32) * NTP_OUT * 512;
  bf16* BpO = (bf16*)(ws + off); off = alignup(off + (size_t)btot_out * sizeof(bf16));
  float* OUT = (float*)(ws + off); off = alignup(off + (size_t)N * NCLASS * sizeof(float));
  (void)ws_size; (void)n_in; (void)out_size;

  const int TB = 256;
  auto blocks1d = [](long long n, int tb) { return (unsigned)((n + tb - 1) / tb); };

  // ---- cast x into concat buffer; pack all weights ----
  k_cast_x<<<blocks1d((long long)N * NFEAT, TB), TB, 0, stream>>>(x, XB, N);
  for (int k = 0; k < NLAYERS; ++k) {
    int fan_in = NFEAT + k * NHID;
    long long total = ((long long)fan_in / 32) * NTP_HID * 512;
    k_pack_w<<<blocks1d(total, TB), TB, 0, stream>>>(Wk[k], NHID, fan_in, NHID,
                                                     NTP_HID, BpH + bofs[k], total);
  }
  k_pack_w<<<blocks1d(btot_out, TB), TB, 0, stream>>>(W_out, NCLASS, CATDIM, NCLASS,
                                                      NTP_OUT, BpO, btot_out);

  // ---- snowball layers ----
  for (int k = 0; k < NLAYERS; ++k) {
    const int fan_in = NFEAT + k * NHID;
    dim3 grid((N + 127) / 128, (NHID + 63) / 64);
    k_gemm_wmma_bf16<<<grid, TB, 0, stream>>>(XB, CATDIM, BpH + bofs[k], NTP_HID,
                                              H, NHID, N, fan_in, NHID);
    k_zero_f32<<<blocks1d((long long)N * NHID, TB), TB, 0, stream>>>(AH, (long long)N * NHID);
    const int F4 = NHID / 4;
    long long total = E * F4;
    k_spmm<<<blocks1d(total, TB), TB, 0, stream>>>(H, NHID, edge_val, edge_src, edge_dst,
                                                   AH, NHID, F4, total);
    k_bias_tanh_append<<<blocks1d((long long)N * NHID, TB), TB, 0, stream>>>(
        AH, bk[k], XB, NFEAT + k * NHID, N);
  }

  // ---- output layer ----
  {
    dim3 grid((N + 127) / 128, (NCLASS + 63) / 64);
    k_gemm_wmma_bf16<<<grid, TB, 0, stream>>>(XB, CATDIM, BpO, NTP_OUT,
                                              H, NCLASS, N, CATDIM, NCLASS);
    k_zero_f32<<<blocks1d((long long)N * NCLASS, TB), TB, 0, stream>>>(OUT, (long long)N * NCLASS);
    const int F4 = NCLASS / 4;
    long long total = E * F4;
    k_spmm<<<blocks1d(total, TB), TB, 0, stream>>>(H, NCLASS, edge_val, edge_src, edge_dst,
                                                   OUT, NCLASS, F4, total);
    k_bias_logsoftmax<<<blocks1d(N, TB), TB, 0, stream>>>(OUT, b_out, (float*)d_out, N);
  }
}
